// EntityMentionAggregation_82849919140500
// MI455X (gfx1250) — matrix operand order repaired and checked by reading, not compiled, not run on hardware
//
#include <hip/hip_runtime.h>

// ---------------- Types ----------------
typedef __attribute__((ext_vector_type(16))) __bf16 v16bf;
typedef __attribute__((ext_vector_type(8)))  float  v8f;
typedef __attribute__((ext_vector_type(4)))  int    v4i;

union Frag { uint4 q[2]; v16bf v; };

#define H 768
#define BATCH 4096
#define ROWS 65536             // BATCH * 16
#define PROJ_ELEMS 50331648ull // ROWS * H
#define W_ELEMS 589824ull      // H * H
#define NEGF (-65504.0f)

// CDNA5 async global->LDS path (ASYNCcnt), with fallback to reg staging.
#if defined(__gfx1250__) && __has_builtin(__builtin_amdgcn_global_load_async_to_lds_b128) && __has_builtin(__builtin_amdgcn_s_wait_asynccnt)
#define USE_ASYNC_LDS 1
#define AS1 __attribute__((address_space(1)))
#define AS3 __attribute__((address_space(3)))
#define ASYNC_CP16(gsrc, ldst) \
  __builtin_amdgcn_global_load_async_to_lds_b128((AS1 v4i*)(gsrc), (AS3 v4i*)(ldst), 0, 0)
#else
#define USE_ASYNC_LDS 0
#endif

// ---------------- helpers ----------------
__device__ __forceinline__ unsigned short f2bf(float f) {
  union { float f; unsigned u; } v; v.f = f;
  unsigned u = v.u;
  unsigned r = (u + 0x7FFFu + ((u >> 16) & 1u)) >> 16;
  if ((u & 0x7F800000u) == 0x7F800000u)
    r = (u >> 16) | (((u & 0xFFFFu) != 0u) ? 0x40u : 0u);
  return (unsigned short)r;
}
__device__ __forceinline__ float bf2f(unsigned short h) {
  union { unsigned u; float f; } v; v.u = ((unsigned)h) << 16; return v.f;
}

// ---------------- K0: f32 -> bf16 convert ----------------
__global__ __launch_bounds__(256)
void cvt_f32_bf16_kernel(const float* __restrict__ src,
                         unsigned short* __restrict__ dst, int n4) {
  int i = blockIdx.x * blockDim.x + threadIdx.x;
  const int stride = gridDim.x * blockDim.x;
  for (; i < n4; i += stride) {
    float4 f = reinterpret_cast<const float4*>(src)[i];
    unsigned r0 = f2bf(f.x), r1 = f2bf(f.y), r2 = f2bf(f.z), r3 = f2bf(f.w);
    uint2 o; o.x = r0 | (r1 << 16); o.y = r2 | (r3 << 16);
    reinterpret_cast<uint2*>(dst)[i] = o;
  }
}

// ---------------- K1: projection GEMM Y = X * W^T + b (bf16 WMMA) ----------------
// Grid: 5 (proj) * 512 (row tiles of 128) * 6 (col tiles of 128). 256 threads.
__global__ __launch_bounds__(256)
void proj_gemm_kernel(const unsigned short* __restrict__ Xh,
                      const unsigned short* __restrict__ Xt,
                      const unsigned short* __restrict__ Wall,
                      const float* __restrict__ b0, const float* __restrict__ b1,
                      const float* __restrict__ b2, const float* __restrict__ b3,
                      const float* __restrict__ b4,
                      unsigned short* __restrict__ Yall) {
  const int blk = blockIdx.x;
  const int p = blk / (512 * 6);
  const int rem = blk % (512 * 6);
  const int rowTile = rem / 6;
  const int colTile = rem % 6;
  const int tid = threadIdx.x;
  const int lane = tid & 31;
  const int wave = tid >> 5;
  const int waveM = wave & 3;   // 4 waves -> 32-row slabs
  const int waveN = wave >> 2;  // 2 waves -> 64-col slabs
  const int rowBase = rowTile * 128;
  const int colBase = colTile * 128;

  const unsigned short* X = (p == 4) ? Xt : Xh;
  const unsigned short* W = Wall + (size_t)p * W_ELEMS;
  const float* bias = (p == 0) ? b0 : (p == 1) ? b1 : (p == 2) ? b2 : (p == 3) ? b3 : b4;
  unsigned short* Y = Yall + (size_t)p * PROJ_ELEMS;

  __shared__ __align__(16) unsigned short lA[2][128][32];
  __shared__ __align__(16) unsigned short lB[2][128][32];

  v8f acc[2][4] = {};

  // staging: 512 uint4 chunks per matrix per step, 2 per thread
  const int c0 = tid, c1 = tid + 256;
  const int ar0 = c0 >> 2, ac0 = (c0 & 3) << 3;
  const int ar1 = c1 >> 2, ac1 = (c1 & 3) << 3;

#if !USE_ASYNC_LDS
  uint4 ra0, ra1, rb0, rb1;
#endif

  // prologue: stage K-slab 0 into buffer 0
#if USE_ASYNC_LDS
  ASYNC_CP16(X + (size_t)(rowBase + ar0) * H + ac0, &lA[0][ar0][ac0]);
  ASYNC_CP16(X + (size_t)(rowBase + ar1) * H + ac1, &lA[0][ar1][ac1]);
  ASYNC_CP16(W + (size_t)(colBase + ar0) * H + ac0, &lB[0][ar0][ac0]);
  ASYNC_CP16(W + (size_t)(colBase + ar1) * H + ac1, &lB[0][ar1][ac1]);
  __builtin_amdgcn_s_wait_asynccnt(0);
#else
  {
    ra0 = *(const uint4*)(X + (size_t)(rowBase + ar0) * H + ac0);
    ra1 = *(const uint4*)(X + (size_t)(rowBase + ar1) * H + ac1);
    rb0 = *(const uint4*)(W + (size_t)(colBase + ar0) * H + ac0);
    rb1 = *(const uint4*)(W + (size_t)(colBase + ar1) * H + ac1);
    *(uint4*)&lA[0][ar0][ac0] = ra0;
    *(uint4*)&lA[0][ar1][ac1] = ra1;
    *(uint4*)&lB[0][ar0][ac0] = rb0;
    *(uint4*)&lB[0][ar1][ac1] = rb1;
  }
#endif
  __syncthreads();

  const int half = (lane >> 4) << 3;  // A: K sub-offset per half-wave
  const int kb   = (lane >> 4) << 4;  // unused; B reads use half, like A

  #pragma unroll 1
  for (int kt = 0; kt < 24; ++kt) {
    const int cur = kt & 1;
    if (kt < 23) {  // stage next K-slab while WMMAs run on current one
      const int k0 = (kt + 1) * 32;
#if USE_ASYNC_LDS
      const int nxt = cur ^ 1;
      ASYNC_CP16(X + (size_t)(rowBase + ar0) * H + k0 + ac0, &lA[nxt][ar0][ac0]);
      ASYNC_CP16(X + (size_t)(rowBase + ar1) * H + k0 + ac1, &lA[nxt][ar1][ac1]);
      ASYNC_CP16(W + (size_t)(colBase + ar0) * H + k0 + ac0, &lB[nxt][ar0][ac0]);
      ASYNC_CP16(W + (size_t)(colBase + ar1) * H + k0 + ac1, &lB[nxt][ar1][ac1]);
#else
      ra0 = *(const uint4*)(X + (size_t)(rowBase + ar0) * H + k0 + ac0);
      ra1 = *(const uint4*)(X + (size_t)(rowBase + ar1) * H + k0 + ac1);
      rb0 = *(const uint4*)(W + (size_t)(colBase + ar0) * H + k0 + ac0);
      rb1 = *(const uint4*)(W + (size_t)(colBase + ar1) * H + k0 + ac1);
#endif
    }
    Frag fa[2], fb[4];
    #pragma unroll
    for (int i = 0; i < 2; ++i) {
      const int mrow = waveM * 32 + i * 16 + (lane & 15);
      fa[i].q[0] = *(const uint4*)&lA[cur][mrow][half];
      fa[i].q[1] = *(const uint4*)&lA[cur][mrow][16 + half];
    }
    #pragma unroll
    for (int j = 0; j < 4; ++j) {
      const int bcol = waveN * 64 + j * 16 + (lane & 15);
      fb[j].q[0] = *(const uint4*)&lB[cur][bcol][half];
      fb[j].q[1] = *(const uint4*)&lB[cur][bcol][16 + half];
    }
    #pragma unroll
    for (int i = 0; i < 2; ++i)
      #pragma unroll
      for (int j = 0; j < 4; ++j)
        acc[i][j] = __builtin_amdgcn_wmma_f32_16x16x32_bf16(
            false, fa[i].v, false, fb[j].v, (short)0, acc[i][j], false, false);
    if (kt < 23) {
#if USE_ASYNC_LDS
      __builtin_amdgcn_s_wait_asynccnt(0);
#else
      const int nxt = cur ^ 1;
      *(uint4*)&lA[nxt][ar0][ac0] = ra0;
      *(uint4*)&lA[nxt][ar1][ac1] = ra1;
      *(uint4*)&lB[nxt][ar0][ac0] = rb0;
      *(uint4*)&lB[nxt][ar1][ac1] = rb1;
#endif
    }
    __syncthreads();
  }

  // epilogue: bias add + bf16 store.  VGPR r -> M = r (+8 for lanes>=16)
  #pragma unroll
  for (int i = 0; i < 2; ++i) {
    #pragma unroll
    for (int j = 0; j < 4; ++j) {
      const int gn = colBase + waveN * 64 + j * 16 + (lane & 15);
      const float bv = bias[gn];
      const int mb = rowBase + waveM * 32 + i * 16 + ((lane >> 4) << 3);
      #pragma unroll
      for (int r = 0; r < 8; ++r)
        Y[(size_t)(mb + r) * H + gn] = f2bf(acc[i][j][r] + bv);
    }
  }
}

// ---------------- K2: per-batch attention + gate + pooling ----------------
// 1 block per batch, 256 threads (8 waves). Waves 0/1 do score WMMAs.
__global__ __launch_bounds__(256)
void attn_fuse_kernel(const unsigned short* __restrict__ proj,   // 5 x PROJ_ELEMS
                      const unsigned short* __restrict__ tailb,  // bf16 tail
                      const int* __restrict__ hmask, const int* __restrict__ tmask,
                      const float* __restrict__ Wg, const float* __restrict__ bgp,
                      float* __restrict__ out) {
  const int b = blockIdx.x;
  const int tid = threadIdx.x;
  const int lane = tid & 31;
  const int wave = tid >> 5;
  const size_t rowOff = (size_t)b * 16 * H;
  const unsigned short* sq = proj + 0 * PROJ_ELEMS + rowOff;
  const unsigned short* sk = proj + 1 * PROJ_ELEMS + rowOff;
  const unsigned short* sv = proj + 2 * PROJ_ELEMS + rowOff;
  const unsigned short* cq = proj + 3 * PROJ_ELEMS + rowOff;
  const unsigned short* ck = proj + 4 * PROJ_ELEMS + rowOff;
  const unsigned short* tl = tailb + rowOff;

  __shared__ float sw[256], cw[256], glog[16], gate[16], mw[16];
  if (tid < 16) glog[tid] = 0.0f;

  if (wave < 2) {
    const unsigned short* Q = wave ? cq : sq;
    const unsigned short* K = wave ? ck : sk;
    const int* msk = wave ? tmask : hmask;
    float* wdst = wave ? cw : sw;
    const int mrow = lane & 15;
    const int half = (lane >> 4) << 3;
    const int kb = (lane >> 4) << 4;
    v8f acc = {};
    #pragma unroll 1
    for (int k0 = 0; k0 < H; k0 += 32) {
      Frag fa, fb;
      fa.q[0] = *(const uint4*)(Q + mrow * H + k0 + half);
      fa.q[1] = *(const uint4*)(Q + mrow * H + k0 + 16 + half);
      fb.q[0] = *(const uint4*)(K + mrow * H + k0 + half);
      fb.q[1] = *(const uint4*)(K + mrow * H + k0 + 16 + half);
      acc = __builtin_amdgcn_wmma_f32_16x16x32_bf16(
          false, fa.v, false, fb.v, (short)0, acc, false, false);
    }
    const float scale = 0.03608439182435161f;  // 1/sqrt(768)
    const int n = lane & 15;
    const int mv = msk[b * 16 + n];
    #pragma unroll
    for (int r = 0; r < 8; ++r) {
      const int m = r + ((lane >> 4) << 3);
      float v = acc[r] * scale;
      if (mv == 0) v = NEGF;
      float mx = v;
      for (int o = 8; o; o >>= 1) mx = fmaxf(mx, __shfl_xor(mx, o, 16));
      float e = __expf(v - mx);
      float s = e;
      for (int o = 8; o; o >>= 1) s += __shfl_xor(s, o, 16);
      wdst[m * 16 + n] = e / s;
    }
  }
  __syncthreads();

  // pass A: gate logits (recompute so/co later; LDS stays tiny)
  {
    float ga[16];
    #pragma unroll
    for (int m = 0; m < 16; ++m) ga[m] = 0.0f;
    for (int c = 0; c < 3; ++c) {
      const int h = tid + c * 256;
      float so[16], co[16];
      #pragma unroll
      for (int m = 0; m < 16; ++m) { so[m] = 0.0f; co[m] = 0.0f; }
      for (int n = 0; n < 16; ++n) {
        const float svv = bf2f(sv[n * H + h]);
        const float tlv = bf2f(tl[n * H + h]);
        #pragma unroll
        for (int m = 0; m < 16; ++m) {
          so[m] += sw[m * 16 + n] * svv;
          co[m] += cw[m * 16 + n] * tlv;
        }
      }
      const float wgs = Wg[h], wgc = Wg[H + h];
      #pragma unroll
      for (int m = 0; m < 16; ++m) ga[m] += so[m] * wgs + co[m] * wgc;
    }
    #pragma unroll
    for (int m = 0; m < 16; ++m) atomicAdd(&glog[m], ga[m]);
  }
  __syncthreads();

  if (tid < 16) {
    gate[tid] = 1.0f / (1.0f + __expf(-(glog[tid] + bgp[0])));
    float s = 0.0f;
    for (int j = 0; j < 16; ++j) s += __expf((float)hmask[b * 16 + j]);
    mw[tid] = __expf((float)hmask[b * 16 + tid]) / s;
  }
  __syncthreads();

  // pass B: fused output + mask-weighted pooling
  for (int c = 0; c < 3; ++c) {
    const int h = tid + c * 256;
    float so[16], co[16];
    #pragma unroll
    for (int m = 0; m < 16; ++m) { so[m] = 0.0f; co[m] = 0.0f; }
    for (int n = 0; n < 16; ++n) {
      const float svv = bf2f(sv[n * H + h]);
      const float tlv = bf2f(tl[n * H + h]);
      #pragma unroll
      for (int m = 0; m < 16; ++m) {
        so[m] += sw[m * 16 + n] * svv;
        co[m] += cw[m * 16 + n] * tlv;
      }
    }
    float a = 0.0f;
    #pragma unroll
    for (int m = 0; m < 16; ++m) {
      const float g = gate[m];
      a += mw[m] * (g * so[m] + (1.0f - g) * co[m]);
    }
    out[(size_t)b * H + h] = a;
  }
}

// ---------------- host launch ----------------
extern "C" void kernel_launch(void* const* d_in, const int* in_sizes, int n_in,
                              void* d_out, int out_size, void* d_ws, size_t ws_size,
                              hipStream_t stream) {
  const float* head = (const float*)d_in[0];
  const float* tail = (const float*)d_in[1];
  const int* hmask = (const int*)d_in[2];
  const int* tmask = (const int*)d_in[3];
  const float* Wsrc[5] = { (const float*)d_in[4], (const float*)d_in[6],
                           (const float*)d_in[8], (const float*)d_in[10],
                           (const float*)d_in[12] };
  const float* b0 = (const float*)d_in[5];
  const float* b1 = (const float*)d_in[7];
  const float* b2 = (const float*)d_in[9];
  const float* b3 = (const float*)d_in[11];
  const float* b4 = (const float*)d_in[13];
  const float* Wg = (const float*)d_in[14];
  const float* bg = (const float*)d_in[15];

  unsigned short* wsH = (unsigned short*)d_ws;          // ROWS*H bf16
  unsigned short* wsT = wsH + PROJ_ELEMS;               // ROWS*H bf16
  unsigned short* wsW = wsT + PROJ_ELEMS;               // 5*H*H bf16
  unsigned short* wsP = wsW + 5 * W_ELEMS;              // 5*ROWS*H bf16
  const size_t need = (2 * PROJ_ELEMS + 5 * W_ELEMS + 5 * PROJ_ELEMS) * sizeof(unsigned short);
  if (ws_size < need) return;

  // K0: conversions
  cvt_f32_bf16_kernel<<<4096, 256, 0, stream>>>(head, wsH, (int)(PROJ_ELEMS >> 2));
  cvt_f32_bf16_kernel<<<4096, 256, 0, stream>>>(tail, wsT, (int)(PROJ_ELEMS >> 2));
  for (int p = 0; p < 5; ++p)
    cvt_f32_bf16_kernel<<<576, 256, 0, stream>>>(Wsrc[p], wsW + (size_t)p * W_ELEMS,
                                                 (int)(W_ELEMS >> 2));

  // K1: five projections
  proj_gemm_kernel<<<5 * 512 * 6, 256, 0, stream>>>(wsH, wsT, wsW, b0, b1, b2, b3, b4, wsP);

  // K2: fused attention / gate / pooling
  attn_fuse_kernel<<<BATCH, 256, 0, stream>>>(wsP, wsT, hmask, tmask, Wg, bg, (float*)d_out);
}
